// Mamba2Block_5351529251298
// MI455X (gfx1250) — matrix operand order repaired and checked by reading, not compiled, not run on hardware
//
#include <hip/hip_runtime.h>

// ---------------- types ----------------
typedef __attribute__((ext_vector_type(16))) __bf16 v16bf;
typedef __attribute__((ext_vector_type(8)))  __bf16 bf16x8;
typedef __attribute__((ext_vector_type(8)))  float  v8f;
typedef __attribute__((ext_vector_type(4)))  unsigned int u32x4;
typedef __attribute__((ext_vector_type(4)))  int i32x4;

#define SHUF16(lo, hh) __builtin_shufflevector(lo, hh, 0,1,2,3,4,5,6,7,8,9,10,11,12,13,14,15)

// ---------------- problem dims ----------------
#define LDW     8256        // fused xz|params row width (4096 + 4160-pad)
#define NROWS   8192        // B*L
#define DMODEL  1024
#define DINNER  2048
#define NHEADS  32
#define DHEAD   64
#define DSTATE  64
#define PDIM    129
#define NCHUNK  16
#define CHUNK   256
#define PCOLB   4096        // params start column in xzp

// ---------------- gfx1250 async-copy support (probe via __has_builtin) ----------------
#if defined(__HIP_DEVICE_COMPILE__) && __has_builtin(__builtin_amdgcn_global_load_async_to_lds_b128)
#define HAS_ASYNC 1
#else
#define HAS_ASYNC 0
#endif

#if defined(__HIP_DEVICE_COMPILE__) && __has_builtin(__builtin_amdgcn_s_wait_asynccnt)
#define WAITA(n) __builtin_amdgcn_s_wait_asynccnt(n)
#else
#define WAITA(n) asm volatile("s_wait_asynccnt %0" :: "i"(n) : "memory")
#endif

#if HAS_ASYNC
__device__ __forceinline__ void async_ld16(const __bf16* g, __bf16* l) {
  __builtin_amdgcn_global_load_async_to_lds_b128(
      (__attribute__((address_space(1))) i32x4*)(g),
      (__attribute__((address_space(3))) i32x4*)(l), 0, 0);
}
#endif

// ---------------- WMMA helpers ----------------
__device__ __forceinline__ v8f wmma_bf16(v16bf a, v16bf b, v8f c) {
  return __builtin_amdgcn_wmma_f32_16x16x32_bf16(false, a, false, b, (short)0, c, false, false);
}

// A fragment: logical 16x32 (MxK), memory row-major with row stride ld (elements).
// lane l -> row = l&15 ; K halves at 8*(l>>4) and 16+8*(l>>4); each half 8 contiguous bf16.
__device__ __forceinline__ v16bf frag_a(const __bf16* base, int ld) {
  int l  = threadIdx.x & 31;
  int m  = l & 15;
  int hi = l >> 4;
  const __bf16* p = base + (size_t)m * ld + 8 * hi;
  bf16x8 lo = *(const bf16x8*)p;
  bf16x8 hh = *(const bf16x8*)(p + 16);
  return SHUF16(lo, hh);
}

// B fragment: logical 32x16 (KxN) with operand stored as N x K row-major (row stride ld).
// lane l -> n = l&15 ; K halves at 8*(l>>4) and 16+8*(l>>4); each half 8 contiguous bf16.
__device__ __forceinline__ v16bf frag_b(const __bf16* base, int ld) {
  int l  = threadIdx.x & 31;
  int n  = l & 15;
  int hi = l >> 4;
  const __bf16* p = base + (size_t)n * ld + 8 * hi;
  bf16x8 lo = *(const bf16x8*)p;
  bf16x8 hh = *(const bf16x8*)(p + 16);
  return SHUF16(lo, hh);
}

__device__ __forceinline__ void wait_ds() {
  asm volatile("s_wait_dscnt 0" ::: "memory");
}

// ---------------- conversion kernels ----------------
__global__ void cvt_f32_bf16(const float* __restrict__ in, __bf16* __restrict__ out, int n) {
  int i = (blockIdx.x * blockDim.x + threadIdx.x) * 4;
  if (i + 3 < n) {
    out[i + 0] = (__bf16)in[i + 0];
    out[i + 1] = (__bf16)in[i + 1];
    out[i + 2] = (__bf16)in[i + 2];
    out[i + 3] = (__bf16)in[i + 3];
  }
}

// Pack W_in (4096x1024) and W_param (4128x1024, padded to 4160 rows of zeros) into one 8256x1024 bf16 matrix.
__global__ void build_wcat(const float* __restrict__ Win, const float* __restrict__ Wpm,
                           __bf16* __restrict__ out) {
  int idx = blockIdx.x * blockDim.x + threadIdx.x;   // over 8256*1024
  if (idx >= LDW * DMODEL) return;
  int r = idx >> 10;
  int c = idx & 1023;
  float v;
  if (r < 4096)       v = Win[(size_t)r * DMODEL + c];
  else if (r < 8224)  v = Wpm[(size_t)(r - 4096) * DMODEL + c];
  else                v = 0.f;
  out[idx] = (__bf16)v;
}

// ---------------- bf16 WMMA GEMM with async-staged B tile ----------------
// C(MxN,f32) = A(MxK) * B^T, B stored NxK row-major.
// Block = 8 waves stacked in M (512 rows) sharing one 64-col B tile staged in LDS,
// double-buffered via the gfx1250 async global->LDS path (ASYNCcnt).
// grid: (N/64, M/512), block 256.
__global__ __launch_bounds__(256) void gemm_bf16_lds(const __bf16* __restrict__ A,
                                                     const __bf16* __restrict__ Bw,
                                                     float* __restrict__ C,
                                                     int K, int lda, int ldb, int ldc) {
  __shared__ __attribute__((aligned(16))) __bf16 Bs[2][64][40];

  const int tn   = blockIdx.x;
  const int bm   = blockIdx.y;
  const int wave = threadIdx.x >> 5;

  const __bf16* Ap = A + (size_t)(bm * 512 + wave * 64) * lda;
  const __bf16* Bp = Bw + (size_t)tn * 64 * ldb;

  // staging map: 256 threads cover 64 rows x 4 x 16B chunks of the 64x32 tile
  const int sr = threadIdx.x >> 2;   // 0..63 : B row (n)
  const int sq = threadIdx.x & 3;    // 0..3  : 8-element chunk in K
  const __bf16* gsrc = Bp + (size_t)sr * ldb + sq * 8;
  __bf16* ldst0 = &Bs[0][sr][sq * 8];
  __bf16* ldst1 = &Bs[1][sr][sq * 8];

  v8f acc[4][4] = {};
  const int nsteps = K >> 5;

#if HAS_ASYNC
  async_ld16(gsrc, ldst0);
#else
  u32x4 pre = *(const u32x4*)gsrc;
#endif

  for (int step = 0; step < nsteps; ++step) {
    const int k0 = step << 5;
#if HAS_ASYNC
    if (step + 1 < nsteps) {
      async_ld16(gsrc + k0 + 32, (step & 1) ? ldst0 : ldst1);
      WAITA(1);
    } else {
      WAITA(0);
    }
#else
    *(u32x4*)((step & 1) ? ldst1 : ldst0) = pre;
    if (step + 1 < nsteps) pre = *(const u32x4*)(gsrc + k0 + 32);
#endif
    __syncthreads();

    const __bf16* Bb = &Bs[step & 1][0][0];
    v16bf bfr[4], af[4];
#pragma unroll
    for (int j = 0; j < 4; ++j) bfr[j] = frag_b(Bb + j * 16 * 40, 40);
#pragma unroll
    for (int i = 0; i < 4; ++i) af[i] = frag_a(Ap + (size_t)i * 16 * lda + k0, lda);
#pragma unroll
    for (int i = 0; i < 4; ++i)
#pragma unroll
      for (int j = 0; j < 4; ++j)
        acc[i][j] = wmma_bf16(af[i], bfr[j], acc[i][j]);

    __syncthreads();
  }

  const int lane = threadIdx.x & 31;
  const int n    = lane & 15;
  const int hi   = lane >> 4;
#pragma unroll
  for (int i = 0; i < 4; ++i)
#pragma unroll
    for (int j = 0; j < 4; ++j)
#pragma unroll
      for (int r = 0; r < 8; ++r) {
        int m = r + 8 * hi;
        C[(size_t)(bm * 512 + wave * 64 + i * 16 + m) * ldc + tn * 64 + j * 16 + n] = acc[i][j][r];
      }
}

// ---------------- SSD phase A: per (chunk, nb) intra-chunk work ----------------
// grid (NCHUNK, 64), block 256 (8 waves)
__global__ __launch_bounds__(256) void ssd_phaseA(const float* __restrict__ xzp,
                                                  float* __restrict__ ylocal,
                                                  float* __restrict__ cumOut,
                                                  float* __restrict__ Tc,
                                                  float* __restrict__ decayOut) {
  __shared__ __attribute__((aligned(16))) __bf16 Bc[256][72];    // (j, n)
  __shared__ __attribute__((aligned(16))) __bf16 Cc[256][72];    // (t, n)
  __shared__ __attribute__((aligned(16))) __bf16 Xt[64][264];    // (p, j)
  __shared__ __attribute__((aligned(16))) __bf16 Btw[64][264];   // (n, j) decay-weighted
  __shared__ __attribute__((aligned(16))) __bf16 bounce[8][16][40]; // per-wave 16x32 relayout strip
  __shared__ float cumS[256];

  const int c  = blockIdx.x;
  const int nb = blockIdx.y;
  const int b  = nb >> 5;
  const int hd = nb & 31;
  const int r0 = b * 4096 + c * CHUNK;
  const int pcol0 = PCOLB + hd * PDIM;
  const int t = threadIdx.x;

  // ---- a -> log(clip(sigmoid)) ----
  {
    float v = xzp[(size_t)(r0 + t) * LDW + pcol0];
    float s = 1.f / (1.f + __expf(-v));
    cumS[t] = __logf(fmaxf(s, 1e-10f));
  }

  // ---- stage B, C, X^T as bf16 ----
  {
    int lane64 = t & 63;
    int rq = t >> 6;
    for (int j = rq; j < 256; j += 4) {
      const float* prow = xzp + (size_t)(r0 + j) * LDW;
      float bv = prow[pcol0 + 1 + lane64];
      float cv = prow[pcol0 + 65 + lane64];
      float xv = prow[hd * 64 + lane64];
      Bc[j][lane64] = (__bf16)bv;
      Cc[j][lane64] = (__bf16)cv;
      Xt[lane64][j] = (__bf16)xv;
    }
  }
  __syncthreads();

  // ---- inclusive scan of log-decay (Hillis-Steele) ----
  for (int off = 1; off < 256; off <<= 1) {
    float add = (t >= off) ? cumS[t - off] : 0.f;
    __syncthreads();
    cumS[t] += add;
    __syncthreads();
  }

  // ---- stage decay-weighted B^T ----
  {
    float cl = cumS[255];
    int lane64 = t & 63;
    int rq = t >> 6;
    for (int j = rq; j < 256; j += 4) {
      float w = __expf(cl - cumS[j]);
      Btw[lane64][j] = (__bf16)(w * (float)Bc[j][lane64]);
    }
  }
  __syncthreads();

  cumOut[(size_t)nb * 4096 + c * CHUNK + t] = cumS[t];
  if (t == 0) decayOut[c * 64 + nb] = __expf(cumS[255]);

  const int wave = t >> 5;
  const int lane = t & 31;
  const int nidx = lane & 15;
  const int hi   = lane >> 4;

  // ---- y_local: each wave handles two 16-row strips (balanced triangular load) ----
  for (int ss = 0; ss < 2; ++ss) {
    int s = ss ? (15 - wave) : wave;
    v8f acc0 = {}, acc1 = {}, acc2 = {}, acc3 = {};
    for (int jt2 = 0; jt2 <= (s >> 1); ++jt2) {
      int j0 = jt2 * 32;
      // cb tiles (16x16 each), K = DSTATE = 64
      v8f c0 = {}, c1 = {};
#pragma unroll
      for (int k = 0; k < 64; k += 32) {
        v16bf afC = frag_a(&Cc[16 * s][k], 72);
        c0 = wmma_bf16(afC, frag_b(&Bc[j0][k], 72), c0);
        c1 = wmma_bf16(afC, frag_b(&Bc[j0 + 16][k], 72), c1);
      }
      // mask with causal decay and bounce through LDS to A-fragment layout
#pragma unroll
      for (int r = 0; r < 8; ++r) {
        int m  = r + 8 * hi;
        int tg = 16 * s + m;
        int jA = j0 + nidx;
        int jB = j0 + 16 + nidx;
        float lA = (tg >= jA) ? __expf(cumS[tg] - cumS[jA]) : 0.f;
        float lB = (tg >= jB) ? __expf(cumS[tg] - cumS[jB]) : 0.f;
        bounce[wave][m][nidx]      = (__bf16)(c0[r] * lA);
        bounce[wave][m][16 + nidx] = (__bf16)(c1[r] * lB);
      }
      wait_ds();  // intra-wave cross-lane LDS RAW
      v16bf am = frag_a(&bounce[wave][0][0], 40);
      acc0 = wmma_bf16(am, frag_b(&Xt[0][j0],  264), acc0);
      acc1 = wmma_bf16(am, frag_b(&Xt[16][j0], 264), acc1);
      acc2 = wmma_bf16(am, frag_b(&Xt[32][j0], 264), acc2);
      acc3 = wmma_bf16(am, frag_b(&Xt[48][j0], 264), acc3);
    }
    // store y_local (b,l,e) layout
#pragma unroll
    for (int r = 0; r < 8; ++r) {
      int m = r + 8 * hi;
      float* yp = ylocal + (size_t)(r0 + 16 * s + m) * DINNER + hd * 64;
      yp[ 0 + nidx] = acc0[r];
      yp[16 + nidx] = acc1[r];
      yp[32 + nidx] = acc2[r];
      yp[48 + nidx] = acc3[r];
    }
  }

  // ---- chunk summary T_c = (w . B)^T @ X  (64x64, K=256): 16 tiles over 8 waves ----
  for (int q = 2 * wave; q < 2 * wave + 2; ++q) {
    int tn = q >> 2;
    int tp = q & 3;
    v8f acc = {};
#pragma unroll
    for (int k = 0; k < 256; k += 32)
      acc = wmma_bf16(frag_a(&Btw[16 * tn][k], 264), frag_b(&Xt[16 * tp][k], 264), acc);
#pragma unroll
    for (int r = 0; r < 8; ++r) {
      int m = r + 8 * hi;
      Tc[(size_t)(c * 64 + nb) * 4096 + (16 * tn + m) * 64 + 16 * tp + nidx] = acc[r];
    }
  }
}

// ---------------- SSD phase B: sequential (cheap) h recurrence ----------------
// grid 64, block 256
__global__ __launch_bounds__(256) void ssd_phaseB(const float* __restrict__ Tc,
                                                  const float* __restrict__ decay,
                                                  const float* __restrict__ h0,
                                                  float* __restrict__ hstates,
                                                  float* __restrict__ hout) {
  int nb = blockIdx.x;
  int t  = threadIdx.x;
  float h[16];
#pragma unroll
  for (int k = 0; k < 16; ++k) h[k] = h0[(size_t)nb * 4096 + k * 256 + t];
  for (int c = 0; c < NCHUNK; ++c) {
    float d = decay[c * 64 + nb];
    size_t base = (size_t)(c * 64 + nb) * 4096;
#pragma unroll
    for (int k = 0; k < 16; ++k) {
      hstates[base + k * 256 + t] = h[k];           // pre-chunk state
      h[k] = d * h[k] + Tc[base + k * 256 + t];
    }
  }
#pragma unroll
  for (int k = 0; k < 16; ++k) hout[(size_t)nb * 4096 + k * 256 + t] = h[k];
}

// ---------------- SSD phase C: y_init + SiLU gate + bf16 pack ----------------
// grid (NCHUNK, 64), block 256
__global__ __launch_bounds__(256) void ssd_phaseC(const float* __restrict__ xzp,
                                                  const float* __restrict__ ylocal,
                                                  const float* __restrict__ cumOut,
                                                  const float* __restrict__ hstates,
                                                  __bf16* __restrict__ ybf) {
  __shared__ float hs[64][65];
  const int c  = blockIdx.x;
  const int nb = blockIdx.y;
  const int b  = nb >> 5;
  const int hd = nb & 31;
  const int r0 = b * 4096 + c * CHUNK;
  const int t  = threadIdx.x;

  for (int k = t; k < 4096; k += 256)
    hs[k >> 6][k & 63] = hstates[(size_t)(c * 64 + nb) * 4096 + k];
  __syncthreads();

  size_t row = (size_t)(r0 + t);
  const float* prow = xzp + row * LDW;
  const int pcol0 = PCOLB + hd * PDIM;

  float yi[64];
#pragma unroll
  for (int p = 0; p < 64; ++p) yi[p] = 0.f;
  for (int n = 0; n < 64; ++n) {
    float cv = prow[pcol0 + 65 + n];
#pragma unroll
    for (int p = 0; p < 64; ++p) yi[p] += cv * hs[n][p];
  }

  float ec = __expf(cumOut[(size_t)nb * 4096 + c * CHUNK + t]);
  const float* ylp = ylocal + row * DINNER + hd * 64;
  const float* zp  = prow + DINNER + hd * 64;
  __bf16* yo = ybf + row * DINNER + hd * 64;
#pragma unroll
  for (int p = 0; p < 64; ++p) {
    float z  = zp[p];
    float sz = z / (1.f + __expf(-z));
    yo[p] = (__bf16)((ylp[p] + ec * yi[p]) * sz);
  }
}

// ---------------- host launcher ----------------
extern "C" void kernel_launch(void* const* d_in, const int* in_sizes, int n_in,
                              void* d_out, int out_size, void* d_ws, size_t ws_size,
                              hipStream_t stream) {
  (void)in_sizes; (void)n_in; (void)out_size; (void)ws_size;
  const float* x    = (const float*)d_in[0];
  const float* h0   = (const float*)d_in[1];
  const float* Win  = (const float*)d_in[2];
  const float* Wpm  = (const float*)d_in[3];
  const float* Wout = (const float*)d_in[4];
  float* out  = (float*)d_out;
  float* hout = out + (size_t)NROWS * DMODEL;   // 8388608

  char* ws = (char*)d_ws;
  size_t off = 0;
  auto take = [&](size_t bytes) -> void* {
    void* p = ws + off;
    off += (bytes + 255) & ~(size_t)255;
    return p;
  };
  __bf16* xbf    = (__bf16*)take((size_t)NROWS * DMODEL * 2);
  __bf16* wcat   = (__bf16*)take((size_t)LDW * DMODEL * 2);
  __bf16* woutbf = (__bf16*)take((size_t)DMODEL * DINNER * 2);
  float*  xzp    = (float*) take((size_t)NROWS * LDW * 4);
  float*  ylocal = (float*) take((size_t)NROWS * DINNER * 4);
  __bf16* ybf    = (__bf16*)take((size_t)NROWS * DINNER * 2);
  float*  cumW   = (float*) take((size_t)64 * 4096 * 4);
  float*  TcW    = (float*) take((size_t)NCHUNK * 64 * 4096 * 4);
  float*  decayW = (float*) take((size_t)NCHUNK * 64 * 4);
  float*  hstW   = (float*) take((size_t)NCHUNK * 64 * 4096 * 4);

  // 1) convert x -> bf16
  {
    int n = NROWS * DMODEL;
    cvt_f32_bf16<<<n / (256 * 4), 256, 0, stream>>>(x, xbf, n);
  }
  // 2) pack W_in|W_param -> bf16 (padded)
  {
    int n = LDW * DMODEL;
    build_wcat<<<(n + 255) / 256, 256, 0, stream>>>(Win, Wpm, wcat);
  }
  // 3) convert W_out -> bf16
  {
    int n = DMODEL * DINNER;
    cvt_f32_bf16<<<n / (256 * 4), 256, 0, stream>>>(Wout, woutbf, n);
  }
  // 4) fused input GEMM: xzp(8192 x 8256) = x(8192x1024) * [W_in|W_param]^T
  gemm_bf16_lds<<<dim3(LDW / 64, NROWS / 512), 256, 0, stream>>>(
      xbf, wcat, xzp, DMODEL, DMODEL, DMODEL, LDW);
  // 5) SSD phases
  ssd_phaseA<<<dim3(NCHUNK, 64), 256, 0, stream>>>(xzp, ylocal, cumW, TcW, decayW);
  ssd_phaseB<<<64, 256, 0, stream>>>(TcW, decayW, h0, hstW, hout);
  ssd_phaseC<<<dim3(NCHUNK, 64), 256, 0, stream>>>(xzp, ylocal, cumW, hstW, ybf);
  // 6) output GEMM: out(8192x1024) = y(8192x2048) * W_out^T
  gemm_bf16_lds<<<dim3(DMODEL / 64, NROWS / 512), 256, 0, stream>>>(
      ybf, woutbf, out, DINNER, DINNER, DINNER, DMODEL);
}
